// SpectralMatching_43490838839885
// MI455X (gfx1250) — matrix operand order repaired and checked by reading, not compiled, not run on hardware
//
#include <hip/hip_runtime.h>
#include <math.h>

// ---------------- problem constants ----------------
constexpr int kN      = 10000;          // NUM_SRC*NUM_DST
constexpr int kIters  = 50;             // MAX_ITER
constexpr int kKP     = 10016;          // K padded to multiple of 32
constexpr int kSteps  = kKP / 32;       // 313 WMMA k-steps
constexpr int kSlices = 8;              // K-slices per row tile (one wave each)
constexpr int kChunk  = (kSteps + kSlices - 1) / kSlices;  // 40
constexpr int kTiles  = kN / 16;        // 625 row tiles
constexpr int kRB     = 80;             // reduction blocks

// ---------------- WMMA fragment types ----------------
typedef __attribute__((ext_vector_type(16))) __bf16 v16bf;
typedef __attribute__((ext_vector_type(8)))  float  v8f;
typedef __attribute__((ext_vector_type(4)))  int    v4i;
typedef __attribute__((ext_vector_type(4)))  __bf16 v4bf;

union FragAB { v16bf v; v4i q[2]; };

// ---------------- kernels ----------------

// Convert f32 M (row-major, pitch kN) -> bf16 Mb (pitch kKP, zero-padded tail).
__global__ __launch_bounds__(256) void convert_kernel(const float* __restrict__ M,
                                                      __bf16* __restrict__ Mb) {
    const int row = blockIdx.x;                       // grid = kN rows
    const float*  src = M  + (size_t)row * kN;
    __bf16*       dst = Mb + (size_t)row * kKP;
    for (int c4 = threadIdx.x * 4; c4 < kKP; c4 += blockDim.x * 4) {
        float4 f = make_float4(0.f, 0.f, 0.f, 0.f);
        if (c4 < kN) f = *(const float4*)(src + c4);  // kN,kKP both /4 -> no straddle
        v4bf o;
        o[0] = (__bf16)f.x; o[1] = (__bf16)f.y; o[2] = (__bf16)f.z; o[3] = (__bf16)f.w;
        *(v4bf*)(dst + c4) = o;
    }
}

// v0 = ones (padding zeroed).  grid*block covers kKP.
__global__ __launch_bounds__(256) void init_kernel(__bf16* __restrict__ vb) {
    int j = blockIdx.x * blockDim.x + threadIdx.x;
    if (j < kKP) vb[j] = (j < kN) ? (__bf16)1.0f : (__bf16)0.0f;
}

// y_part[slice][row] = sum over this wave's K-slice of M[row][j]*v[j].
// block = 8 waves = 8 K-slices of one 16-row tile; grid = kTiles.
// A (16x32) = vb chunk broadcast across M-rows; B (32x16) = M tile (row-major,
// each lane reads 32 contiguous bytes).  C accumulates in f32.
template <bool F32SRC>
__global__ __launch_bounds__(256) void matvec_kernel(const void* __restrict__ Msrc,
                                                     const __bf16* __restrict__ vb,
                                                     float* __restrict__ y_part) {
    const int lane  = threadIdx.x & 31;
    // Wave-uniform values -> force into SGPRs so loop control is scalar
    // (s_cmp/s_cbranch, EXEC untouched around the WMMAs).
    const int slice = __builtin_amdgcn_readfirstlane(threadIdx.x >> 5);
    const int tile  = blockIdx.x;
    const int n     = lane & 15;
    const int hi    = lane >> 4;           // lane half selects K sub-range
    const int row   = tile * 16 + n;

    int step        = slice * kChunk;
    int stepE       = step + kChunk; if (stepE > kSteps) stepE = kSteps;
    const int pfClamp = stepE - 1;         // last valid step (prefetch clamp)

    const char* aBase = (const char*)vb + 32 * hi;   // K += 16 for upper half
    const char* bRow;
    if (F32SRC) bRow = (const char*)Msrc + (size_t)row * ((size_t)kN * 4) + (size_t)64 * hi;
    else        bRow = (const char*)Msrc + (size_t)row * ((size_t)kKP * 2) + (size_t)32 * hi;

    auto doStep = [&](int st, v8f cacc) -> v8f {
        const int jb = st * 32;
        FragAB a, b;
        // A: vb[jb + hi*16 .. +15]  (broadcast across rows)
        const char* ap = aBase + (size_t)jb * 2;
        a.q[0] = *(const v4i*)ap;
        a.q[1] = *(const v4i*)(ap + 16);
        if constexpr (!F32SRC) {
            // B: 32 contiguous bytes of this lane's M row
            const char* bp = bRow + (size_t)jb * 2;
            b.q[0] = *(const v4i*)bp;
            b.q[1] = *(const v4i*)(bp + 16);
            // Branchless prefetch: clamp keeps the address inside this slice's
            // range of Mb (always mapped), so WGP-scope (near) prefetch is safe.
            int pf = st + 8; if (pf > pfClamp) pf = pfClamp;   // scalar s_min
            __builtin_prefetch(bRow + (size_t)pf * 64, 0, 3);
        } else {
            const char* bp = bRow + (size_t)jb * 4;
            if (jb + hi * 16 < kN) {                  // K-tail mask (step 312, upper half)
                float f[16];
                *(float4*)(f + 0)  = *(const float4*)(bp + 0);
                *(float4*)(f + 4)  = *(const float4*)(bp + 16);
                *(float4*)(f + 8)  = *(const float4*)(bp + 32);
                *(float4*)(f + 12) = *(const float4*)(bp + 48);
#pragma unroll
                for (int e = 0; e < 16; ++e) b.v[e] = (__bf16)f[e];
            } else {
                b.q[0] = (v4i){0, 0, 0, 0};
                b.q[1] = (v4i){0, 0, 0, 0};
            }
        }
        return __builtin_amdgcn_wmma_f32_16x16x32_bf16(false, a.v, false, b.v,
                                                       (short)0, cacc, false, false);
    };

    v8f c0 = {}; v8f c1 = {};
    for (; step + 1 < stepE; step += 2) {   // 2 accumulators: break WMMA RAW chain
        c0 = doStep(step, c0);
        c1 = doStep(step + 1, c1);
    }
    if (step < stepE) c0 = doStep(step, c0);
    v8f c = c0 + c1;

    // All C rows identical; lane n's c[0] = partial dot for output row.
    if (lane < 16) y_part[slice * kN + row] = c[0];
}

// Per-block partial sums of ||y||^2, y_j = sum over slices of y_part.
__global__ __launch_bounds__(256) void sq_kernel(const float* __restrict__ y_part,
                                                 float* __restrict__ sblk) {
    float local = 0.f;
    for (int j = blockIdx.x * blockDim.x + threadIdx.x; j < kN; j += gridDim.x * blockDim.x) {
        float yj = 0.f;
#pragma unroll
        for (int sl = 0; sl < kSlices; ++sl) yj += y_part[sl * kN + j];
        local += yj * yj;
    }
#pragma unroll
    for (int off = 16; off > 0; off >>= 1) local += __shfl_xor(local, off);
    __shared__ float wsum[8];
    if ((threadIdx.x & 31) == 0) wsum[threadIdx.x >> 5] = local;
    __syncthreads();
    if (threadIdx.x == 0) {
        float t = 0.f;
        for (int i = 0; i < 8; ++i) t += wsum[i];
        sblk[blockIdx.x] = t;
    }
}

// v = y/||y|| -> bf16 for next iteration; f32 to d_out on last iteration.
__global__ __launch_bounds__(256) void norm_kernel(const float* __restrict__ y_part,
                                                   const float* __restrict__ sblk,
                                                   __bf16* __restrict__ vb,
                                                   float* __restrict__ out, int last) {
    float ssum = 0.f;
    for (int i = 0; i < kRB; ++i) ssum += sblk[i];   // fixed order -> deterministic
    const float t = 1.0f / sqrtf(ssum);
    for (int j = blockIdx.x * blockDim.x + threadIdx.x; j < kN; j += gridDim.x * blockDim.x) {
        float yj = 0.f;
#pragma unroll
        for (int sl = 0; sl < kSlices; ++sl) yj += y_part[sl * kN + j];
        const float vj = yj * t;
        vb[j] = (__bf16)vj;
        if (last) out[j] = vj;
    }
}

// ---------------- host launcher ----------------
extern "C" void kernel_launch(void* const* d_in, const int* in_sizes, int n_in,
                              void* d_out, int out_size, void* d_ws, size_t ws_size,
                              hipStream_t stream) {
    (void)in_sizes; (void)n_in; (void)out_size;
    const float* M = (const float*)d_in[0];
    char* ws = (char*)d_ws;

    auto alignUp = [](size_t x, size_t a) { return (x + a - 1) & ~(a - 1); };
    size_t off = 0;
    const size_t off_y  = off; off += (size_t)kSlices * kN * sizeof(float); off = alignUp(off, 256);
    const size_t off_vb = off; off += (size_t)kKP * 2;                      off = alignUp(off, 256);
    const size_t off_sb = off; off += (size_t)kRB * sizeof(float);          off = alignUp(off, 256);
    const size_t off_m  = off;
    const size_t sz_m   = (size_t)kN * kKP * 2;     // ~191 MiB bf16 matrix (fits global L2)
    const size_t need_min  = off_m;
    const size_t need_full = off_m + sz_m;

    if (ws_size < need_min) return;                 // cannot run safely

    float*  y_part = (float*)(ws + off_y);
    __bf16* vb     = (__bf16*)(ws + off_vb);
    float*  sblk   = (float*)(ws + off_sb);
    __bf16* Mb     = (__bf16*)(ws + off_m);
    float*  out    = (float*)d_out;

    const bool bf16path = (ws_size >= need_full);

    if (bf16path)
        convert_kernel<<<kN, 256, 0, stream>>>(M, Mb);     // one ~0.6GB pass, halves per-iter traffic
    init_kernel<<<(kKP + 255) / 256, 256, 0, stream>>>(vb);

    for (int k = 0; k < kIters; ++k) {
        if (bf16path)
            matvec_kernel<false><<<kTiles, 256, 0, stream>>>((const void*)Mb, vb, y_part);
        else
            matvec_kernel<true><<<kTiles, 256, 0, stream>>>((const void*)M, vb, y_part);
        sq_kernel<<<kRB, 256, 0, stream>>>(y_part, sblk);
        norm_kernel<<<kRB, 256, 0, stream>>>(y_part, sblk, vb, out, k == kIters - 1 ? 1 : 0);
    }
}
